// HomeostaticNCA_13580686590114
// MI455X (gfx1250) — matrix operand order repaired and checked by reading, not compiled, not run on hardware
//
#include <hip/hip_runtime.h>
#include <hip/hip_bf16.h>
#include <math.h>

typedef __attribute__((ext_vector_type(16))) _Float16 v16h;
typedef __attribute__((ext_vector_type(8)))  _Float16 v8h;
typedef __attribute__((ext_vector_type(8)))  float    v8f;

#define N_CH   16
#define N_MEM  8
#define N_HW   4
#define N_HEADS 4
#define TOT    28
#define PERCF  112
#define HH     256
#define WW     256
#define BB     8

// ---------------- WMMA fragment helpers (CDNA5 16x16x32 f16, wave32) ----------------

// A fragment (16x32, f16) from LDS row-major buf[16][ldk].
// Layout: lanes 0-15: M=lane, K = k0+{0..15}; lanes 16-31: M=lane-16, K = k0+{16..31} (matches load_b).
__device__ inline v16h load_a(const _Float16* buf, int ldk, int k0, int lane) {
    int row = lane & 15;
    int kb  = k0 + ((lane >> 4) & 1) * 16;
    const _Float16* r = buf + row * ldk + kb;
    v8h lo = *(const v8h*)(r);        // K = kb .. kb+7   (16B aligned)
    v8h hi = *(const v8h*)(r + 8);    // K = kb+8 .. kb+15
    v16h a;
#pragma unroll
    for (int h = 0; h < 8; ++h) { a[h] = lo[h]; a[h + 8] = hi[h]; }
    return a;
}

// B fragment (32x16, f16) from global weights stored row-major [N][K] (f16).
// Lane holds column N = n0+(lane&15); K = k0 + (lane<16 ? 0..15 : 16..31) contiguous -> one 32B load.
__device__ inline v16h load_b(const _Float16* w, int ldk, int n0, int k0, int lane) {
    int col = lane & 15;
    int kb  = k0 + ((lane >> 4) & 1) * 16;
    return *(const v16h*)(w + (size_t)(n0 + col) * ldk + kb);
}

// C/D fragment: lane holds N = n0+(lane&15); VGPR j holds M = j + (lane<16 ? 0 : 8).
__device__ inline v8f bias_frag(const float* bias, int n0, int lane) {
    float bv = bias[n0 + (lane & 15)];
    v8f c;
#pragma unroll
    for (int j = 0; j < 8; ++j) c[j] = bv;
    return c;
}

__device__ inline void store_c_h(_Float16* buf, int ldk, int n0, v8f c, int lane, bool do_relu) {
    int col = lane & 15;
    int rb  = ((lane >> 4) & 1) * 8;
#pragma unroll
    for (int j = 0; j < 8; ++j) {
        float v = c[j];
        if (do_relu) v = v > 0.f ? v : 0.f;
        buf[(rb + j) * ldk + n0 + col] = (_Float16)v;
    }
}

__device__ inline v8f wmma_f16(v16h a, v16h b, v8f c) {
    return __builtin_amdgcn_wmma_f32_16x16x32_f16(false, a, false, b, (short)0, c, false, false);
}

// ---------------- kernel 0: convert weights to f16 [N][K] (padded) in workspace ----------------

__global__ __launch_bounds__(256) void nca_init(const float* __restrict__ W1, const float* __restrict__ W2,
                                                const float* __restrict__ pw1, const float* __restrict__ pw2,
                                                _Float16* __restrict__ w1h, _Float16* __restrict__ w2h,
                                                _Float16* __restrict__ pw1h, _Float16* __restrict__ pw2h,
                                                float* __restrict__ cnt) {
    int i = blockIdx.x * 256 + threadIdx.x;
    if (i < 8) cnt[i] = 0.f;
    int j = i;
    if (j < 128 * 128) {                       // W1 (128,112) -> [128][128] K-padded
        int n = j >> 7, k = j & 127;
        w1h[j] = (_Float16)((k < PERCF) ? W1[n * PERCF + k] : 0.f);
        return;
    }
    j -= 128 * 128;
    if (j < 64 * 128) { w2h[j] = (_Float16)W2[j]; return; }     // W2 (64,128) -> [64][128]
    j -= 64 * 128;
    if (j < 128 * 64) { pw1h[j] = (_Float16)pw1[j]; return; }   // pw1 (4,32,64) -> [128][64]
    j -= 128 * 64;
    if (j < 4 * 32 * 32) {                      // pw2 (4,24,32) -> [4][32][32] N-padded
        int k = j >> 10, r = j & 1023, n = r >> 5, h = r & 31;
        pw2h[j] = (_Float16)((n < N_CH + N_MEM) ? pw2[(k * (N_CH + N_MEM) + n) * 32 + h] : 0.f);
        return;
    }
}

// ---------------- kernel 1: alive count per batch ----------------

__global__ __launch_bounds__(256) void nca_alive(const float* __restrict__ x, float* __restrict__ cnt) {
    __shared__ float red[256];
    int b = blockIdx.x;
    const float* plane = x + ((size_t)(b * TOT + 3) * HH) * WW;
    float s = 0.f;
    for (int i = threadIdx.x; i < HH * WW; i += 256) s += (plane[i] > 0.1f) ? 1.f : 0.f;
    red[threadIdx.x] = s;
    __syncthreads();
    for (int off = 128; off > 0; off >>= 1) {
        if ((int)threadIdx.x < off) red[threadIdx.x] += red[threadIdx.x + off];
        __syncthreads();
    }
    if (threadIdx.x == 0) cnt[b] = red[0];
}

// ---------------- kernel 2: fused NCA step ----------------
// 256 threads = 8 waves; each wave owns one 16-pixel tile (one image row segment).

__global__ __launch_bounds__(256) void nca_main(
    const float* __restrict__ x,
    const float* __restrict__ b1, const float* __restrict__ b2,
    const float* __restrict__ attW, const float* __restrict__ attb, const float* __restrict__ att_temp,
    const float* __restrict__ pb1, const float* __restrict__ pb2,
    const float* __restrict__ hW1, const float* __restrict__ hb1,
    const float* __restrict__ hW2, const float* __restrict__ hb2,
    const float* __restrict__ hW3, const float* __restrict__ hb3,
    const _Float16* __restrict__ w1h, const _Float16* __restrict__ w2h,
    const _Float16* __restrict__ pw1h, const _Float16* __restrict__ pw2h,
    const float* __restrict__ aliveCnt,
    float* __restrict__ out)
{
    __shared__ __align__(32) _Float16 s_act[8][16 * 128];   // per-wave activation staging (f16)
    __shared__ float s_attn[8][16][4];                      // attn[k] * hs per pixel
    __shared__ float s_alive[8][16];                        // alive_ext per pixel

    const int lane = threadIdx.x & 31;
    const int wv   = threadIdx.x >> 5;
    const int tile = blockIdx.x * 8 + wv;      // 32768 tiles total
    const int b    = tile >> 12;               // 4096 tiles per image
    const int t2   = tile & 4095;
    const int y    = t2 >> 4;
    const int x0   = (t2 & 15) << 4;

    _Float16* buf = s_act[wv];

    // ---- perception: 4 stencils x 28 channels -> buf[pixel][0..111], pad 112..127 ----
    {
        const int p  = lane & 15;
        const int gx = x0 + p;
        for (int c0 = 0; c0 < TOT; c0 += 2) {
            int c = c0 + ((lane >> 4) & 1);
            const float* plane = x + ((size_t)(b * TOT + c) * HH) * WW;
            float v[3][3];
#pragma unroll
            for (int dy = 0; dy < 3; ++dy) {
                int yy = y + dy - 1;
#pragma unroll
                for (int dx = 0; dx < 3; ++dx) {
                    int xx = gx + dx - 1;
                    v[dy][dx] = (yy >= 0 && yy < HH && xx >= 0 && xx < WW) ? plane[yy * WW + xx] : 0.f;
                }
            }
            float ident = v[1][1];
            float sx = -v[0][0] + v[0][2] - 2.f * v[1][0] + 2.f * v[1][2] - v[2][0] + v[2][2];
            float sy = -v[0][0] - 2.f * v[0][1] - v[0][2] + v[2][0] + 2.f * v[2][1] + v[2][2];
            float sum = v[0][0] + v[0][1] + v[0][2] + v[1][0] + v[1][1] + v[1][2] + v[2][0] + v[2][1] + v[2][2];
            float lap = sum - 9.f * ident;
            buf[p * 128 + 0 * TOT + c] = (_Float16)ident;
            buf[p * 128 + 1 * TOT + c] = (_Float16)sx;
            buf[p * 128 + 2 * TOT + c] = (_Float16)sy;
            buf[p * 128 + 3 * TOT + c] = (_Float16)lap;
        }
        for (int i = lane; i < 16 * 16; i += 32)
            buf[(i >> 4) * 128 + PERCF + (i & 15)] = (_Float16)0.f;
    }

    // ---- scalar per-pixel: attention softmax, homeostasis MLP, alive_ext, hw passthrough ----
    if (lane < 16) {
        const int p = lane;
        const int gx = x0 + p;
        const size_t base = (size_t)b * TOT * HH * WW + (size_t)y * WW + gx;
        float hw[4];
#pragma unroll
        for (int j = 0; j < 4; ++j) hw[j] = x[base + (size_t)(N_CH + N_MEM + j) * HH * WW];

        float temp = att_temp[0];
        float lg[4], mx = -1e30f;
#pragma unroll
        for (int k = 0; k < 4; ++k) {
            float s = attb[k];
#pragma unroll
            for (int j = 0; j < 4; ++j) s += attW[k * 4 + j] * hw[j];
            lg[k] = s / temp;
            mx = fmaxf(mx, lg[k]);
        }
        float se = 0.f;
#pragma unroll
        for (int k = 0; k < 4; ++k) { lg[k] = expf(lg[k] - mx); se += lg[k]; }

        float mm = 0.f;
#pragma unroll
        for (int m = 0; m < N_MEM; ++m) mm += x[base + (size_t)(N_CH + m) * HH * WW];
        mm *= (1.f / N_MEM);
        float ar = aliveCnt[b] * (1.f / (HH * WW));

        float t1[16];
#pragma unroll
        for (int i = 0; i < 16; ++i) {
            float s = hW1[i * 2 + 0] * mm + hW1[i * 2 + 1] * ar + hb1[i];
            t1[i] = s > 0.f ? s : 0.f;
        }
        float t2v[8];
#pragma unroll
        for (int i = 0; i < 8; ++i) {
            float s = hb2[i];
#pragma unroll
            for (int j = 0; j < 16; ++j) s += hW2[i * 16 + j] * t1[j];
            t2v[i] = s > 0.f ? s : 0.f;
        }
        float z = hb3[0];
#pragma unroll
        for (int j = 0; j < 8; ++j) z += hW3[j] * t2v[j];
        float hs = 1.f / (1.f + expf(-z));

        float inv = hs / se;   // fold hs into attention weights
#pragma unroll
        for (int k = 0; k < 4; ++k) s_attn[wv][p][k] = lg[k] * inv;

        // alive_ext: 3x3 max of (x[b,3] > 0.1)
        const float* ap = x + ((size_t)(b * TOT + 3) * HH) * WW;
        float ae = 0.f;
#pragma unroll
        for (int dy = 0; dy < 3; ++dy) {
            int yy = y + dy - 1;
#pragma unroll
            for (int dx = 0; dx < 3; ++dx) {
                int xx = gx + dx - 1;
                if (yy >= 0 && yy < HH && xx >= 0 && xx < WW && ap[yy * WW + xx] > 0.1f) ae = 1.f;
            }
        }
        s_alive[wv][p] = ae;

        // hardware channels pass through unchanged
#pragma unroll
        for (int j = 0; j < 4; ++j) out[base + (size_t)(N_CH + N_MEM + j) * HH * WW] = hw[j];
    }
    __syncthreads();   // also guarantees full EXEC reconvergence before WMMA

    // ---- L1: 112(pad128) -> 128, ReLU ----
    {
        v16h a0 = load_a(buf, 128, 0, lane), a1 = load_a(buf, 128, 32, lane);
        v16h a2 = load_a(buf, 128, 64, lane), a3 = load_a(buf, 128, 96, lane);
#pragma unroll
        for (int nt = 0; nt < 8; ++nt) {
            v8f c = bias_frag(b1, nt * 16, lane);
            c = wmma_f16(a0, load_b(w1h, 128, nt * 16, 0,  lane), c);
            c = wmma_f16(a1, load_b(w1h, 128, nt * 16, 32, lane), c);
            c = wmma_f16(a2, load_b(w1h, 128, nt * 16, 64, lane), c);
            c = wmma_f16(a3, load_b(w1h, 128, nt * 16, 96, lane), c);
            store_c_h(buf, 128, nt * 16, c, lane, true);   // A is in regs, safe to overwrite LDS
        }
    }
    // ---- L2: 128 -> 64 (proc), no ReLU ----
    {
        v16h a0 = load_a(buf, 128, 0, lane), a1 = load_a(buf, 128, 32, lane);
        v16h a2 = load_a(buf, 128, 64, lane), a3 = load_a(buf, 128, 96, lane);
#pragma unroll
        for (int nt = 0; nt < 4; ++nt) {
            v8f c = bias_frag(b2, nt * 16, lane);
            c = wmma_f16(a0, load_b(w2h, 128, nt * 16, 0,  lane), c);
            c = wmma_f16(a1, load_b(w2h, 128, nt * 16, 32, lane), c);
            c = wmma_f16(a2, load_b(w2h, 128, nt * 16, 64, lane), c);
            c = wmma_f16(a3, load_b(w2h, 128, nt * 16, 96, lane), c);
            store_c_h(buf, 128, nt * 16, c, lane, false);
        }
    }
    // ---- L3: 64 -> 128 (4 heads x 32), ReLU ----
    {
        v16h a0 = load_a(buf, 128, 0, lane), a1 = load_a(buf, 128, 32, lane);
#pragma unroll
        for (int nt = 0; nt < 8; ++nt) {
            v8f c = bias_frag(pb1, nt * 16, lane);   // pb1 is (4,32) = flat 128, head-major == n
            c = wmma_f16(a0, load_b(pw1h, 64, nt * 16, 0,  lane), c);
            c = wmma_f16(a1, load_b(pw1h, 64, nt * 16, 32, lane), c);
            store_c_h(buf, 128, nt * 16, c, lane, true);
        }
    }
    // ---- L4: block-diagonal 4x(32 -> 24) + attention/hs combine ----
    const int col = lane & 15;
    const int rb  = ((lane >> 4) & 1) * 8;
    v8f upd0; v8f upd1;
#pragma unroll
    for (int j = 0; j < 8; ++j) { upd0[j] = 0.f; upd1[j] = 0.f; }
    {
        v16h hh[4];
#pragma unroll
        for (int k = 0; k < 4; ++k) hh[k] = load_a(buf, 128, k * 32, lane);
#pragma unroll
        for (int k = 0; k < 4; ++k) {
#pragma unroll
            for (int nt = 0; nt < 2; ++nt) {
                int cch = nt * 16 + col;
                float bv = (cch < N_CH + N_MEM) ? pb2[k * (N_CH + N_MEM) + cch] : 0.f;
                v8f c;
#pragma unroll
                for (int j = 0; j < 8; ++j) c[j] = bv;
                c = wmma_f16(hh[k], load_b(pw2h + k * 32 * 32, 32, nt * 16, 0, lane), c);
#pragma unroll
                for (int j = 0; j < 8; ++j) {
                    float s = s_attn[wv][rb + j][k];
                    if (nt == 0) upd0[j] += c[j] * s; else upd1[j] += c[j] * s;
                }
            }
        }
    }

    // ---- residual add + alive gate + writeout (channels 0..23) ----
#pragma unroll
    for (int nt = 0; nt < 2; ++nt) {
        int cch = nt * 16 + col;
        if (cch < N_CH + N_MEM) {
#pragma unroll
            for (int j = 0; j < 8; ++j) {
                int p = rb + j;
                int gx = x0 + p;
                size_t idx = ((size_t)(b * TOT + cch) * HH + y) * WW + gx;
                float u = (nt == 0) ? upd0[j] : upd1[j];
                out[idx] = (x[idx] + u) * s_alive[wv][p];
            }
        }
    }
}

// ---------------- launch ----------------

extern "C" void kernel_launch(void* const* d_in, const int* in_sizes, int n_in,
                              void* d_out, int out_size, void* d_ws, size_t ws_size,
                              hipStream_t stream) {
    const float* x        = (const float*)d_in[0];
    const float* W1       = (const float*)d_in[1];
    const float* b1       = (const float*)d_in[2];
    const float* W2       = (const float*)d_in[3];
    const float* b2       = (const float*)d_in[4];
    const float* attW     = (const float*)d_in[5];
    const float* attb     = (const float*)d_in[6];
    const float* att_temp = (const float*)d_in[7];
    const float* pw1      = (const float*)d_in[8];
    const float* pb1      = (const float*)d_in[9];
    const float* pw2      = (const float*)d_in[10];
    const float* pb2      = (const float*)d_in[11];
    const float* hW1      = (const float*)d_in[12];
    const float* hb1      = (const float*)d_in[13];
    const float* hW2      = (const float*)d_in[14];
    const float* hb2      = (const float*)d_in[15];
    const float* hW3      = (const float*)d_in[16];
    const float* hb3      = (const float*)d_in[17];
    float* out = (float*)d_out;

    char* ws = (char*)d_ws;
    float*     cnt  = (float*)(ws + 0);
    _Float16*  w1h  = (_Float16*)(ws + 64);
    _Float16*  w2h  = (_Float16*)(ws + 64 + 32768);
    _Float16*  pw1h = (_Float16*)(ws + 64 + 32768 + 16384);
    _Float16*  pw2h = (_Float16*)(ws + 64 + 32768 + 16384 + 16384);

    nca_init<<<144, 256, 0, stream>>>(W1, W2, pw1, pw2, w1h, w2h, pw1h, pw2h, cnt);
    nca_alive<<<BB, 256, 0, stream>>>(x, cnt);

    // 32768 tiles of 16 pixels; 8 waves (tiles) per 256-thread block -> 4096 blocks
    nca_main<<<4096, 256, 0, stream>>>(x, b1, b2, attW, attb, att_temp, pb1, pb2,
                                       hW1, hb1, hW2, hb2, hW3, hb3,
                                       w1h, w2h, pw1h, pw2h, cnt, out);
}
